// LocalOTLoss_11261404250190
// MI455X (gfx1250) — matrix operand order repaired and checked by reading, not compiled, not run on hardware
//
#include <hip/hip_runtime.h>
#include <math.h>

// ---------- problem constants (from reference setup_inputs) ----------
#define B_    512
#define NV_   256
#define NT_   128
#define D_    512
#define EPS_  0.1f
#define ITERS_ 5
#define NEGINF_ (-1000000.0f)

typedef __attribute__((ext_vector_type(16))) _Float16 v16h;
typedef __attribute__((ext_vector_type(8)))  _Float16 v8h;
typedef __attribute__((ext_vector_type(4)))  _Float16 h4;
typedef __attribute__((ext_vector_type(8)))  float    v8f;

// ---------- Tensor Data Mover availability ----------
#ifdef __has_builtin
#  if __has_builtin(__builtin_amdgcn_tensor_load_to_lds) && \
      __has_builtin(__builtin_amdgcn_s_wait_tensorcnt)
#    define USE_TDM 1
#  endif
#endif
#ifndef USE_TDM
#  define USE_TDM 0
#endif

#if USE_TDM
typedef unsigned int u32x4 __attribute__((ext_vector_type(4)));
typedef int          i32x4 __attribute__((ext_vector_type(4)));
typedef int          i32x8 __attribute__((ext_vector_type(8)));

// Issue one TDM 2D-tile load: tile = rows x 32 f16 elements, global row
// stride = D_ (512) elements, landing contiguously (row-major, 32-elem rows)
// at LDS byte offset lds_off.  D# bitfields per CDNA5 ISA ch.8 (groups 0/1).
__device__ __forceinline__ void tdm_load_panel(const _Float16* gsrc,
                                               unsigned lds_off, int rows) {
  const unsigned long long ga = (unsigned long long)(uintptr_t)gsrc;
  u32x4 g0;
  g0[0] = 1u;                                            // count=1 (valid user D#)
  g0[1] = lds_off;                                       // lds_addr (bytes)
  g0[2] = (unsigned)(ga & 0xFFFFFFFFull);                // global_addr[31:0]
  g0[3] = (unsigned)((ga >> 32) & 0x01FFFFFFull)         // global_addr[56:32]
          | 0x80000000u;                                 // type=2 ("image")
  i32x8 g1;
  g1[0] = 1 << 16;                 // workgroup_mask=0 | data_size=1 (2 bytes)
  g1[1] = (int)(512u << 16);       // atomic_barrier=0 | tensor_dim0[15:0]=512
  g1[2] = rows << 16;              // tensor_dim0[31:16]=0 | tensor_dim1[15:0]=rows
  g1[3] = 32 << 16;                // tensor_dim1[31:16]=0 | tile_dim0=32
  g1[4] = rows;                    // tile_dim1=rows | tile_dim2=0
  g1[5] = 512;                     // tensor_dim0_stride[31:0] = 512 elems
  g1[6] = 0;                       // stride hi | tensor_dim1_stride lo
  g1[7] = 0;
  i32x4 z4 = {0, 0, 0, 0};
#if __clang_major__ >= 23
  i32x8 z8 = {0, 0, 0, 0, 0, 0, 0, 0};
  __builtin_amdgcn_tensor_load_to_lds(g0, g1, z4, z4, z8, 0);
#else
  __builtin_amdgcn_tensor_load_to_lds(g0, g1, z4, z4, 0);
#endif
}
#endif  // USE_TDM

// ---------------------------------------------------------------------
// Kernel 0: zero the scalar output (atomicAdd target)
// ---------------------------------------------------------------------
__global__ void k_zero(float* out) { out[0] = 0.0f; }

// ---------------------------------------------------------------------
// Kernel 1: row-wise L2 normalize, emit f16 copies of v and t.
// One wave32 per 512-float row: 16 floats/lane via float4 loads.
// ---------------------------------------------------------------------
__global__ __launch_bounds__(256)
void k_normalize(const float* __restrict__ v, const float* __restrict__ t,
                 _Float16* __restrict__ vh, _Float16* __restrict__ th) {
  const int wave = threadIdx.x >> 5;
  const int lane = threadIdx.x & 31;
  const long long row    = (long long)blockIdx.x * 8 + wave;
  const long long nvRows = (long long)B_ * NV_;
  const long long total  = nvRows + (long long)B_ * NT_;
  if (row >= total) return;

  const float* src;
  _Float16*    dst;
  if (row < nvRows) { src = v + row * (long long)D_;            dst = vh + row * (long long)D_; }
  else { long long r2 = row - nvRows; src = t + r2 * (long long)D_; dst = th + r2 * (long long)D_; }

  const float4* s4 = (const float4*)src;
  float4 f[4];
  float ss = 0.0f;
#pragma unroll
  for (int j = 0; j < 4; ++j) {
    f[j] = s4[j * 32 + lane];
    ss += f[j].x * f[j].x + f[j].y * f[j].y + f[j].z * f[j].z + f[j].w * f[j].w;
  }
#pragma unroll
  for (int off = 16; off > 0; off >>= 1) ss += __shfl_xor(ss, off, 32);

  const float inv = 1.0f / fmaxf(sqrtf(ss), 1e-12f);
  h4* d4 = (h4*)dst;
#pragma unroll
  for (int j = 0; j < 4; ++j) {
    h4 hv;
    hv.x = (_Float16)(f[j].x * inv);
    hv.y = (_Float16)(f[j].y * inv);
    hv.z = (_Float16)(f[j].z * inv);
    hv.w = (_Float16)(f[j].w * inv);
    d4[j * 32 + lane] = hv;
  }
}

// ---------------------------------------------------------------------
// Kernel 2: per-batch cosine matrix A[b] = vh[b] (256x512) * th[b]^T (512x128)
// 512 threads = 16 waves, arranged 8 (M) x 2 (N). Wave tile: 32x64 ->
// 2x4 accumulators of 16x16 f32. K streamed in chunks of 32 through
// double-buffered LDS (TDM-staged when available); 8 WMMAs per wave/step.
// ---------------------------------------------------------------------
__global__ __launch_bounds__(512)
void k_gemm_cos(const _Float16* __restrict__ vh, const _Float16* __restrict__ th,
                float* __restrict__ Aout) {
  __shared__ __attribute__((aligned(32))) _Float16 Al[2][NV_ * 32]; // 2 x 16KB
  __shared__ __attribute__((aligned(32))) _Float16 Bl[2][NT_ * 32]; // 2 x 8KB

  const int b      = blockIdx.x;
  const int tid    = threadIdx.x;
  const int wave   = tid >> 5;
  const int lane   = tid & 31;
  const int laneLo = lane & 15;
  const int laneHi = lane >> 4;          // 0 or 1
  const int mbase  = (wave >> 1) * 32;   // 8 M-groups of 32 rows
  const int nbase  = (wave & 1) * 64;    // 2 N-groups of 64 cols
  const int STEPS  = D_ / 32;

  const _Float16* vb = vh + (size_t)b * NV_ * D_;
  const _Float16* tb = th + (size_t)b * NT_ * D_;

  v8f acc[2][4];
#pragma unroll
  for (int mt = 0; mt < 2; ++mt)
#pragma unroll
    for (int nt = 0; nt < 4; ++nt)
#pragma unroll
      for (int e = 0; e < 8; ++e) acc[mt][nt][e] = 0.0f;

#if USE_TDM
  // Low 32 bits of the flat shared-aperture address == LDS byte offset.
  const unsigned aOff[2] = { (unsigned)(uintptr_t)(void*)&Al[0][0],
                             (unsigned)(uintptr_t)(void*)&Al[1][0] };
  const unsigned bOff[2] = { (unsigned)(uintptr_t)(void*)&Bl[0][0],
                             (unsigned)(uintptr_t)(void*)&Bl[1][0] };
  if (wave == 0) {                      // one wave drives the DMA engine
    tdm_load_panel(vb, aOff[0], NV_);
    tdm_load_panel(tb, bOff[0], NT_);
  }
#else
  auto stage = [&](int k0, int buf) {
    // A panel: 256 rows x 32 halfs = 1024 16B copies -> 2 per thread
#pragma unroll
    for (int it = 0; it < 2; ++it) {
      const int idx = tid + it * 512;
      const int row = idx >> 2, c8 = idx & 3;
      *(v8h*)&Al[buf][row * 32 + c8 * 8] =
          *(const v8h*)&vb[(size_t)row * D_ + k0 + c8 * 8];
    }
    // B panel: 128 rows x 32 halfs = 512 16B copies -> 1 per thread
    {
      const int row = tid >> 2, c8 = tid & 3;
      *(v8h*)&Bl[buf][row * 32 + c8 * 8] =
          *(const v8h*)&tb[(size_t)row * D_ + k0 + c8 * 8];
    }
  };
  stage(0, 0);
#endif

  for (int ks = 0; ks < STEPS; ++ks) {
#if USE_TDM
    if (wave == 0) __builtin_amdgcn_s_wait_tensorcnt(0); // buf[ks&1] landed
    __syncthreads();                    // data visible; compute(ks-1) done
    if (wave == 0 && ks + 1 < STEPS) {  // overlap next DMA with compute
      tdm_load_panel(vb + (ks + 1) * 32, aOff[(ks + 1) & 1], NV_);
      tdm_load_panel(tb + (ks + 1) * 32, bOff[(ks + 1) & 1], NT_);
    }
#else
    __syncthreads();                    // staging(ks) visible; compute(ks-1) done
    if (ks + 1 < STEPS) stage((ks + 1) * 32, (ks + 1) & 1);
#endif
    const int buf = ks & 1;

    // A fragments: 16x32 f16, lanes 0-15 = rows, laneHi selects K {0..7,16..23} vs {8..15,24..31}
    v16h af[2];
#pragma unroll
    for (int mt = 0; mt < 2; ++mt) {
      const _Float16* p = &Al[buf][(mbase + mt * 16 + laneLo) * 32 + laneHi * 8];
      v8h lo = *(const v8h*)p;
      v8h hi = *(const v8h*)(p + 16);
#pragma unroll
      for (int e = 0; e < 8; ++e) { af[mt][e] = lo[e]; af[mt][8 + e] = hi[e]; }
    }
    // B fragments: 32x16 f16 with B[k][n] = t[n][k]; lane n, laneHi selects K {0..7,16..23} vs {8..15,24..31}
    v16h bf[4];
#pragma unroll
    for (int nt = 0; nt < 4; ++nt) {
      const _Float16* p = &Bl[buf][(nbase + nt * 16 + laneLo) * 32 + laneHi * 8];
      v8h lo = *(const v8h*)p;
      v8h hi = *(const v8h*)(p + 16);
#pragma unroll
      for (int e = 0; e < 8; ++e) { bf[nt][e] = lo[e]; bf[nt][8 + e] = hi[e]; }
    }

#pragma unroll
    for (int mt = 0; mt < 2; ++mt)
#pragma unroll
      for (int nt = 0; nt < 4; ++nt)
        acc[mt][nt] = __builtin_amdgcn_wmma_f32_16x16x32_f16(
            false, af[mt], false, bf[nt], (short)0, acc[mt][nt], false, false);
  }

  // Epilogue: C/D layout -> element (m,n): lane = (n%16) + 16*(m>=8), vgpr r = m%8
  float* Ab = Aout + (size_t)b * NV_ * NT_;
#pragma unroll
  for (int mt = 0; mt < 2; ++mt)
#pragma unroll
    for (int nt = 0; nt < 4; ++nt)
#pragma unroll
      for (int r = 0; r < 8; ++r) {
        const int m = mbase + mt * 16 + laneHi * 8 + r;
        const int n = nbase + nt * 16 + laneLo;
        Ab[(size_t)m * NT_ + n] = acc[mt][nt][r];
      }
}

// ---------------------------------------------------------------------
// Kernel 3: per-batch Sinkhorn entirely in LDS.
// Augmented K matrix: (NV+1) x (NT+1) = 257x129 f32 = 132.6 KB < 320 KB.
// 5 iterations of stabilized logsumexp u/w sweeps, then transport loss.
// ---------------------------------------------------------------------
__global__ __launch_bounds__(512)
void k_sinkhorn(const float* __restrict__ Aw,
                const unsigned char* __restrict__ vmask,
                const unsigned char* __restrict__ tmask,
                const float* __restrict__ gammap,
                float* __restrict__ out) {
  extern __shared__ float smem[];
  float* Ks  = smem;                 // 257*129
  float* u   = Ks + 257 * 129;       // 257
  float* w   = u + 257;              // 129
  float* red = w + 129;              // 16 wave partials
  float* cnt = red + 16;             // 2 counts

  const int b   = blockIdx.x;
  const int tid = threadIdx.x;
  const float gamma = gammap[0];
  const float invE  = 1.0f / EPS_;
  const float* Ab = Aw + (size_t)b * NV_ * NT_;
  const unsigned char* vm = vmask + (size_t)b * NV_;
  const unsigned char* tm = tmask + (size_t)b * NT_;

  // Build K = masked(A)/eps with gamma borders; init w = 0.
  for (int i = tid; i < NV_ * NT_; i += 512) {
    const int n = i >> 7, m = i & (NT_ - 1);
    const float a  = Ab[i];
    const float am = (vm[n] && tm[m]) ? a : NEGINF_;
    Ks[n * 129 + m] = am * invE;
  }
  for (int n = tid; n < NV_; n += 512) Ks[n * 129 + 128] = gamma * invE;
  for (int m = tid; m < 129; m += 512) { Ks[NV_ * 129 + m] = gamma * invE; w[m] = 0.0f; }
  if (tid == 0) {
    float vc = 0.0f, tc = 0.0f;
    for (int n = 0; n < NV_; ++n) vc += vm[n] ? 1.0f : 0.0f;
    for (int m = 0; m < NT_; ++m) tc += tm[m] ? 1.0f : 0.0f;
    cnt[0] = vc + 1e-9f;
    cnt[1] = tc + 1e-9f;
  }
  __syncthreads();
  const float vcnt = cnt[0], tcnt = cnt[1];

  for (int it = 0; it < ITERS_; ++it) {
    // u[n] = log_mu[n] - logsumexp_m(K[n][m] + w[m])
    for (int n = tid; n < 257; n += 512) {
      const float* kr = Ks + n * 129;
      float mx = -1e30f;
      for (int m = 0; m < 129; ++m) mx = fmaxf(mx, kr[m] + w[m]);
      float s = 0.0f;
      for (int m = 0; m < 129; ++m) s += __expf(kr[m] + w[m] - mx);
      const float lm = (n < NV_) ? __logf((vm[n] ? 1.0f / vcnt : 0.0f) + 1e-9f)
                                 : __logf(1.0f + 1e-9f);
      u[n] = lm - (mx + __logf(s));
    }
    __syncthreads();
    // w[m] = log_nu[m] - logsumexp_n(K[n][m] + u[n])
    for (int m = tid; m < 129; m += 512) {
      float mx = -1e30f;
      for (int n = 0; n < 257; ++n) mx = fmaxf(mx, Ks[n * 129 + m] + u[n]);
      float s = 0.0f;
      for (int n = 0; n < 257; ++n) s += __expf(Ks[n * 129 + m] + u[n] - mx);
      const float ln = (m < NT_) ? __logf((tm[m] ? 1.0f / tcnt : 0.0f) + 1e-9f)
                                 : __logf(1.0f + 1e-9f);
      w[m] = ln - (mx + __logf(s));
    }
    __syncthreads();
  }

  // loss_b = sum over real region of exp(u+w+K) * (1 - A_raw)
  float local = 0.0f;
  for (int i = tid; i < NV_ * NT_; i += 512) {
    const int n = i >> 7, m = i & (NT_ - 1);
    const float T = __expf(u[n] + w[m] + Ks[n * 129 + m]);
    local += T * (1.0f - Ab[i]);
  }
#pragma unroll
  for (int off = 16; off > 0; off >>= 1) local += __shfl_xor(local, off, 32);
  if ((tid & 31) == 0) red[tid >> 5] = local;
  __syncthreads();
  if (tid < 32) {
    float s = (tid < 16) ? red[tid] : 0.0f;
#pragma unroll
    for (int off = 8; off > 0; off >>= 1) s += __shfl_xor(s, off, 32);
    if (tid == 0) atomicAdd(out, s * (1.0f / (float)B_));
  }
}

// ---------------------------------------------------------------------
// Host launcher
// ---------------------------------------------------------------------
extern "C" void kernel_launch(void* const* d_in, const int* in_sizes, int n_in,
                              void* d_out, int out_size, void* d_ws, size_t ws_size,
                              hipStream_t stream) {
  (void)in_sizes; (void)n_in; (void)out_size; (void)ws_size;
  const float*         v  = (const float*)d_in[0];
  const float*         t  = (const float*)d_in[1];
  const unsigned char* vm = (const unsigned char*)d_in[2];
  const unsigned char* tm = (const unsigned char*)d_in[3];
  const float*         g  = (const float*)d_in[4];
  float* out = (float*)d_out;

  // Workspace layout: vh (128MB f16) | th (64MB f16) | A (64MB f32)
  _Float16* vh = (_Float16*)d_ws;
  _Float16* th = vh + (size_t)B_ * NV_ * D_;
  float*    Aw = (float*)(th + (size_t)B_ * NT_ * D_);

  k_zero<<<1, 1, 0, stream>>>(out);

  const long long rows = (long long)B_ * (NV_ + NT_);
  k_normalize<<<(int)((rows + 7) / 8), 256, 0, stream>>>(v, t, vh, th);

  k_gemm_cos<<<B_, 512, 0, stream>>>(vh, th, Aw);

  const size_t sm = (size_t)(257 * 129 + 257 + 129 + 16 + 2) * sizeof(float);
  k_sinkhorn<<<B_, 512, sm, stream>>>(Aw, vm, tm, g, out);
}
